// S5R_84164179132543
// MI455X (gfx1250) — matrix operand order repaired and checked by reading, not compiled, not run on hardware
//
#include <hip/hip_runtime.h>
#include <hip/hip_bf16.h>
#include <math.h>

// ---------------------------------------------------------------------------
// S5 diagonal SSM layer for MI455X (gfx1250, wave32, WMMA, bf16 state).
//   K1 : Bu = B_bar @ u  -> packed bf16 (re,im) words, time-major (B,L,P)
//   K2a: per-chunk aggregates (read-only pass)
//   K2b: sequential carry scan over 16 chunk aggregates (Lambda^512)
//   K2c: chunk re-scan with carry init -> x packed bf16 words
//   K3 : y = Re(C @ x) + diag(D)*u -> exact GELU
// ---------------------------------------------------------------------------

typedef __attribute__((ext_vector_type(16))) __bf16 v16bf;
typedef __attribute__((ext_vector_type(8)))  float  v8f;

#define BB 8
#define HH 256
#define PP 256
#define LL 8192
#define LC 512          // scan chunk length
#define NC (LL / LC)    // 16 chunks

union PkBF2 {           // (re,im) bf16 pair packed in one 32-bit word
    unsigned u;
    __bf16   h[2];
};

__device__ __forceinline__ float gelu_exact(float x) {
    return 0.5f * x * (1.0f + erff(x * 0.70710678118654752f));
}

// A-fragment K index for 16x32 bf16 A tile (contiguous, matches B):
//   element j of lane half kh holds K = 16*kh + j
__device__ __forceinline__ int a_kidx(int j, int kh) {
    return j + (kh << 4);
}

// ---------------------------------------------------------------------------
// K1: Bu(b)(P,L) = B_bar(P,H) x u(b)(H,L), bf16 WMMA.
// One 16(M=p) x 32(N=t) tile per wave (A fragments reused across both N
// tiles), K loop over H in steps of 32. Output packed (re,im) bf16 words.
// ---------------------------------------------------------------------------
__global__ __launch_bounds__(256)
void s5_bu_wmma(const float* __restrict__ u,    // (B,H,L) f32
                const float* __restrict__ Bc,   // (P,H) complex interleaved f32
                unsigned* __restrict__ buw) {   // (B,L,P) packed bf16 pairs
    const int wave = threadIdx.x >> 5;
    const int lane = threadIdx.x & 31;
    const int tile = blockIdx.x * 8 + wave;

    const int TT = LL / 32;                       // 256 time tiles (32 wide)
    const int PT = PP / 16;                       // 16 state tiles
    const int tt = tile % TT;
    const int pt = (tile / TT) % PT;
    const int b  = tile / (TT * PT);

    const int t0 = tt * 32;
    const int p0 = pt * 16;
    const int m  = lane & 15;                     // A-row (p)
    const int n  = lane & 15;                     // B/C col (t)
    const int kh = lane >> 4;                     // K-half select

    v8f accR0 = {}, accI0 = {};                   // t0 .. t0+15
    v8f accR1 = {}, accI1 = {};                   // t0+16 .. t0+31

    const size_t ub = (size_t)b * HH * LL;

    for (int k0 = 0; k0 < HH; k0 += 32) {
        if (k0 + 32 < HH)  // stream hint for the next u K-panel
            __builtin_prefetch(&u[ub + (size_t)(k0 + 32 + kh * 16) * LL + t0], 0, 0);

        v16bf aR, aI, bu0, bu1;
#pragma unroll
        for (int j = 0; j < 16; ++j) {
            const int ka = a_kidx(j, kh);
            const size_t ia = (size_t)(p0 + m) * HH + (k0 + ka);
            aR[j] = (__bf16)Bc[2 * ia];
            aI[j] = (__bf16)Bc[2 * ia + 1];
            const int kb = j + (kh << 4);         // B 32x16: lane half -> K 0-15 / 16-31
            const size_t ibase = ub + (size_t)(k0 + kb) * LL + t0 + n;
            bu0[j] = (__bf16)u[ibase];
            bu1[j] = (__bf16)u[ibase + 16];
        }
        accR0 = __builtin_amdgcn_wmma_f32_16x16x32_bf16(false, aR, false, bu0,
                                                        (short)0, accR0, false, false);
        accI0 = __builtin_amdgcn_wmma_f32_16x16x32_bf16(false, aI, false, bu0,
                                                        (short)0, accI0, false, false);
        accR1 = __builtin_amdgcn_wmma_f32_16x16x32_bf16(false, aR, false, bu1,
                                                        (short)0, accR1, false, false);
        accI1 = __builtin_amdgcn_wmma_f32_16x16x32_bf16(false, aI, false, bu1,
                                                        (short)0, accI1, false, false);
    }

    // C tile: lane holds N=n, VGPR v holds M = v + 8*kh -> 8 contiguous p's.
    // Pack (re,im) into interleaved bf16 pairs -> one 32B store per tile.
    v16bf pk0, pk1;
#pragma unroll
    for (int v = 0; v < 8; ++v) {
        pk0[2 * v]     = (__bf16)accR0[v];
        pk0[2 * v + 1] = (__bf16)accI0[v];
        pk1[2 * v]     = (__bf16)accR1[v];
        pk1[2 * v + 1] = (__bf16)accI1[v];
    }
    const size_t w0 = ((size_t)b * LL + (t0 + n)) * PP + p0 + (kh << 3);
    const size_t w1 = w0 + (size_t)16 * PP;
    *(v16bf*)(buw + w0) = pk0;
    *(v16bf*)(buw + w1) = pk1;
}

// ---------------------------------------------------------------------------
// K2a: per-chunk aggregates (read-only). Block=(b,chunk); thread=p. Coalesced.
//      agg = scan of chunk from zero state (no writeback).
// ---------------------------------------------------------------------------
__global__ __launch_bounds__(256)
void s5_chunk_agg(const unsigned* __restrict__ buw,        // (B,L,P) packed
                  const float* __restrict__ lam,           // (P) complex
                  float* __restrict__ aggR, float* __restrict__ aggI) {
    const int c = blockIdx.x % NC;
    const int b = blockIdx.x / NC;
    const int p = threadIdx.x;

    const float lr = lam[2 * p], li = lam[2 * p + 1];
    float sr = 0.0f, si = 0.0f;

    const size_t base = ((size_t)b * LL + (size_t)c * LC) * PP + p;
    for (int s = 0; s < LC; ++s) {
        PkBF2 w; w.u = buw[base + (size_t)s * PP];
        const float br = (float)w.h[0];
        const float bi = (float)w.h[1];
        const float nr = lr * sr - li * si + br;
        const float ni = lr * si + li * sr + bi;
        sr = nr; si = ni;
    }
    aggR[(b * NC + c) * PP + p] = sr;
    aggI[(b * NC + c) * PP + p] = si;
}

// ---------------------------------------------------------------------------
// K2b: exclusive scan over the NC chunk aggregates per (b,p).
//      Lambda^LC via 9 complex squarings (LC = 2^9).
// ---------------------------------------------------------------------------
__global__ __launch_bounds__(256)
void s5_carry_scan(const float* __restrict__ lam,
                   const float* __restrict__ aggR, const float* __restrict__ aggI,
                   float* __restrict__ carR, float* __restrict__ carI) {
    const int t = blockIdx.x * 256 + threadIdx.x;   // B*P threads
    const int b = t / PP, p = t % PP;

    float ar = lam[2 * p], ai = lam[2 * p + 1];
#pragma unroll
    for (int i = 0; i < 9; ++i) {                   // ar+ai*I = Lambda^512
        const float nr = ar * ar - ai * ai;
        const float ni = 2.0f * ar * ai;
        ar = nr; ai = ni;
    }

    float Sr = 0.0f, Si = 0.0f;                     // state entering chunk c
    for (int c = 0; c < NC; ++c) {
        const int idx = (b * NC + c) * PP + p;
        carR[idx] = Sr;
        carI[idx] = Si;
        const float gr = aggR[idx], gi = aggI[idx];
        const float nr = ar * Sr - ai * Si + gr;
        const float ni = ar * Si + ai * Sr + gi;
        Sr = nr; Si = ni;
    }
}

// ---------------------------------------------------------------------------
// K2c: chunk re-scan initialized with its carry; emit final x in packed bf16.
// ---------------------------------------------------------------------------
__global__ __launch_bounds__(256)
void s5_chunk_scan(const unsigned* __restrict__ buw,       // (B,L,P) packed
                   const float* __restrict__ lam,
                   const float* __restrict__ carR, const float* __restrict__ carI,
                   unsigned* __restrict__ xw) {            // (B,L,P) packed
    const int c = blockIdx.x % NC;
    const int b = blockIdx.x / NC;
    const int p = threadIdx.x;

    const float lr = lam[2 * p], li = lam[2 * p + 1];
    float sr = carR[(b * NC + c) * PP + p];
    float si = carI[(b * NC + c) * PP + p];

    const size_t base = ((size_t)b * LL + (size_t)c * LC) * PP + p;
    for (int s = 0; s < LC; ++s) {
        const size_t idx = base + (size_t)s * PP;
        PkBF2 w; w.u = buw[idx];
        const float nr = lr * sr - li * si + (float)w.h[0];
        const float ni = lr * si + li * sr + (float)w.h[1];
        sr = nr; si = ni;
        PkBF2 o; o.h[0] = (__bf16)sr; o.h[1] = (__bf16)si;
        xw[idx] = o.u;
    }
}

// ---------------------------------------------------------------------------
// K3: y = Re(C @ x) + diag(D)*u -> exact GELU.  16(h) x 32(t) tile per wave,
//     C fragments reused across both N tiles; x loaded as packed bf16.
// ---------------------------------------------------------------------------
__global__ __launch_bounds__(256)
void s5_readout_wmma(const unsigned* __restrict__ xw,  // (B,L,P) packed bf16
                     const float* __restrict__ Cc,     // (H,P) complex interleaved
                     const float* __restrict__ u,      // (B,H,L)
                     const float* __restrict__ D,      // (H,H)
                     float* __restrict__ out) {        // (B,H,L)
    const int wave = threadIdx.x >> 5;
    const int lane = threadIdx.x & 31;
    const int tile = blockIdx.x * 8 + wave;

    const int TT = LL / 32;                        // 256
    const int HT = HH / 16;                        // 16
    const int tt = tile % TT;
    const int ht = (tile / TT) % HT;
    const int b  = tile / (TT * HT);

    const int t0 = tt * 32;
    const int h0 = ht * 16;
    const int m  = lane & 15;
    const int n  = lane & 15;
    const int kh = lane >> 4;

    v8f accR0 = {}, accI0 = {};
    v8f accR1 = {}, accI1 = {};

    for (int k0 = 0; k0 < PP; k0 += 32) {
        v16bf cR, cI, vr0, vi0, vr1, vi1;
#pragma unroll
        for (int j = 0; j < 16; ++j) {
            const int ka = a_kidx(j, kh);
            const size_t ia = (size_t)(h0 + m) * PP + (k0 + ka);
            cR[j] = (__bf16)Cc[2 * ia];
            cI[j] = (__bf16)Cc[2 * ia + 1];
            const int kb = j + (kh << 4);
            const size_t iw0 = ((size_t)b * LL + (t0 + n)) * PP + (k0 + kb);
            PkBF2 w0; w0.u = xw[iw0];
            PkBF2 w1; w1.u = xw[iw0 + (size_t)16 * PP];
            vr0[j] = w0.h[0]; vi0[j] = w0.h[1];
            vr1[j] = w1.h[0]; vi1[j] = w1.h[1];
        }
        accR0 = __builtin_amdgcn_wmma_f32_16x16x32_bf16(false, cR, false, vr0,
                                                        (short)0, accR0, false, false);
        accI0 = __builtin_amdgcn_wmma_f32_16x16x32_bf16(false, cI, false, vi0,
                                                        (short)0, accI0, false, false);
        accR1 = __builtin_amdgcn_wmma_f32_16x16x32_bf16(false, cR, false, vr1,
                                                        (short)0, accR1, false, false);
        accI1 = __builtin_amdgcn_wmma_f32_16x16x32_bf16(false, cI, false, vi1,
                                                        (short)0, accI1, false, false);
    }

#pragma unroll
    for (int v = 0; v < 8; ++v) {
        const int h = h0 + v + (kh << 3);
        const float dh = D[h * HH + h];
        const size_t iu0 = (size_t)b * HH * LL + (size_t)h * LL + (t0 + n);
        const float y0 = accR0[v] - accI0[v] + dh * u[iu0];
        const float y1 = accR1[v] - accI1[v] + dh * u[iu0 + 16];
        out[iu0]      = gelu_exact(y0);
        out[iu0 + 16] = gelu_exact(y1);
    }
}

// ---------------------------------------------------------------------------
extern "C" void kernel_launch(void* const* d_in, const int* in_sizes, int n_in,
                              void* d_out, int out_size, void* d_ws, size_t ws_size,
                              hipStream_t stream) {
    const float* u   = (const float*)d_in[0];   // (B,H,L) f32
    const float* lam = (const float*)d_in[1];   // (P,1) complex64 -> 2P floats
    const float* Bc  = (const float*)d_in[2];   // (P,H) complex64
    const float* Cc  = (const float*)d_in[3];   // (H,P) complex64
    const float* Dm  = (const float*)d_in[4];   // (H,H) f32
    float* out = (float*)d_out;

    const size_t nX = (size_t)BB * LL * PP;     // packed words per state buffer
    const size_t nA = (size_t)BB * NC * PP;     // 32768 aggregates
    unsigned* buw = (unsigned*)d_ws;            // Bu packed bf16 pairs
    unsigned* xw  = buw + nX;                   // x packed bf16 pairs
    float* aggR = (float*)(xw + nX);
    float* aggI = aggR + nA;
    float* carR = aggI + nA;
    float* carI = carR + nA;

    const int tilesK1 = (PP / 16) * (LL / 32) * BB;   // 32768 wave-tiles
    const int tilesK3 = (HH / 16) * (LL / 32) * BB;   // 32768 wave-tiles

    s5_bu_wmma<<<tilesK1 / 8, 256, 0, stream>>>(u, Bc, buw);
    s5_chunk_agg<<<BB * NC, 256, 0, stream>>>(buw, lam, aggR, aggI);
    s5_carry_scan<<<(BB * PP) / 256, 256, 0, stream>>>(lam, aggR, aggI, carR, carI);
    s5_chunk_scan<<<BB * NC, 256, 0, stream>>>(buw, lam, carR, carI, xw);
    s5_readout_wmma<<<tilesK3 / 8, 256, 0, stream>>>(xw, Cc, u, Dm, out);
}
